// GraphConv_77816217469321
// MI455X (gfx1250) — matrix operand order repaired and checked by reading, not compiled, not run on hardware
//
#include <hip/hip_runtime.h>

#define NN 50000
#define EE 1000000
#define EA (EE + NN)
#define GG 1024
#define HH 4
#define MT 3125            // NN/16 exactly

typedef __bf16 bf16;
typedef __attribute__((ext_vector_type(16))) __bf16 v16bf;
typedef __attribute__((ext_vector_type(8)))  __bf16 v8bf;
typedef __attribute__((ext_vector_type(8)))  float  v8f;

static __device__ __forceinline__ unsigned fenc(float f) {
  unsigned u = __float_as_uint(f);
  return (u & 0x80000000u) ? ~u : (u | 0x80000000u);
}
static __device__ __forceinline__ float fdec(unsigned e) {
  unsigned u = (e & 0x80000000u) ? (e ^ 0x80000000u) : ~e;
  return __uint_as_float(u);
}
#define ENC_NEG_INF 0x007FFFFFu   // fenc(-inf)

// ---------------- utility fills ----------------
__global__ void fill_f32(float* p, float v, int n) {
  int i = blockIdx.x * blockDim.x + threadIdx.x;
  if (i < n) p[i] = v;
}
__global__ void fill_u32(unsigned* p, unsigned v, int n) {
  int i = blockIdx.x * blockDim.x + threadIdx.x;
  if (i < n) p[i] = v;
}

// ---------------- self-loop attr (PyG fill_value='mean') ----------------
__global__ void deg_kernel(const int* __restrict__ ei, const float* __restrict__ eattr,
                           float* __restrict__ deg, float* __restrict__ asum) {
  int e = blockIdx.x * blockDim.x + threadIdx.x;
  if (e < EE) {
    int d = ei[EE + e];
    atomicAdd(&deg[d], 1.0f);
    atomicAdd(&asum[d], eattr[e]);
  }
}
__global__ void loop_fin_kernel(float* __restrict__ la, const float* __restrict__ deg) {
  int n = blockIdx.x * blockDim.x + threadIdx.x;
  if (n < NN) la[n] = la[n] / fmaxf(deg[n], 1.0f);
}

// ---------------- input embedding: h0 = embed_w[x] + relu(pos_z*fc_w + fc_b) ----------------
__global__ void embed_kernel(const int* __restrict__ x, const float* __restrict__ pos,
                             const float* __restrict__ ew, const float* __restrict__ fw,
                             const float* __restrict__ fb, bf16* __restrict__ hb) {
  int i = blockIdx.x * blockDim.x + threadIdx.x;
  if (i < NN * 64) {
    int n = i >> 6, j = i & 63;
    float p = pos[n * 3 + 2];
    float v = ew[x[n] * 64 + j] + fmaxf(p * fw[j] + fb[j], 0.0f);
    hb[i] = (bf16)v;            // layer-1 A operand, stride 64
  }
}

// ---------------- pack weight matrix W[cin, hc] (f32) into WMMA B-operand lane layout ----
// B tile (32x16 bf16 per K-step): lane<16 holds K 0..7,16..23 of col=lane; lane>=16 holds K 8..15,24..31.
// Packed contiguous per lane: Bp[((tn*kblocks + kb)*32 + lane)*16 + j]
__global__ void packB_kernel(const float* __restrict__ W, bf16* __restrict__ Bp,
                             int cin, int hc) {
  int kblocks = cin >> 5;
  int total = (hc >> 4) * kblocks * 512;
  int t = blockIdx.x * blockDim.x + threadIdx.x;
  if (t < total) {
    int j = t & 15;
    int lane = (t >> 4) & 31;
    int blk = t >> 9;
    int kb = blk % kblocks;
    int tn = blk / kblocks;
    int col = tn * 16 + (lane & 15);
    int k = kb * 32 + ((lane >> 4) << 3) + (j & 7) + ((j >> 3) << 4);
    Bp[t] = (bf16)W[k * hc + col];
  }
}

// ---------------- per-head edge-attr scalar: kE[h] = dot(linE[h*C: ], att_e[h*C: ]) --------
__global__ void ke_kernel(const float* __restrict__ linE, const float* __restrict__ attE,
                          float* __restrict__ kE, int c) {
  int h = threadIdx.x;
  if (h < HH) {
    float s = 0.0f;
    for (int i = 0; i < c; ++i) s += linE[h * c + i] * attE[h * c + i];
    kE[h] = s;
  }
}

// ---------------- WMMA GEMM: D[NN,hc] = A[NN,cin](bf16) x Bp(packed bf16), f32 accum -------
// One wave per 16x16 output tile; cin and hc are multiples of 32/16; NN = 3125*16.
__global__ __launch_bounds__(256) void gemm_wmma(const bf16* __restrict__ A,
                                                 const bf16* __restrict__ Bp,
                                                 float* __restrict__ D,
                                                 int cin, int hc) {
  int ntc = hc >> 4;
  int wave = (blockIdx.x * blockDim.x + threadIdx.x) >> 5;
  if (wave >= MT * ntc) return;               // uniform per wave: EXEC stays all-1s
  int mt = wave % MT;
  int tn = wave / MT;
  int l = threadIdx.x & 31;
  int row = mt * 16 + (l & 15);
  int khalf = l >> 4;                          // ISA 16-bit A layout: lane-half selects K group
  int kblocks = cin >> 5;
  v8f acc = {};
  for (int kb = 0; kb < kblocks; ++kb) {
    const bf16* ap = A + row * cin + kb * 32 + khalf * 8;
    v8bf alo = *(const v8bf*)(ap);            // K +0..7   (hi lanes: +8..15)
    v8bf ahi = *(const v8bf*)(ap + 16);       // K +16..23 (hi lanes: +24..31)
    v16bf a;
#pragma unroll
    for (int i = 0; i < 8; ++i) { a[i] = alo[i]; a[i + 8] = ahi[i]; }
    v16bf b = *(const v16bf*)(Bp + (((size_t)(tn * kblocks + kb)) * 32 + l) * 16);
    acc = __builtin_amdgcn_wmma_f32_16x16x32_bf16(false, a, false, b,
                                                  (short)0, acc, false, false);
  }
  int mbase = mt * 16 + (khalf << 3);          // C/D layout: lanes>=16 hold M rows 8..15
  int col = tn * 16 + (l & 15);
#pragma unroll
  for (int r = 0; r < 8; ++r) D[(size_t)(mbase + r) * hc + col] = acc[r];
}

// ---------------- per-node attention dots ----------------
__global__ void node_att_kernel(const float* __restrict__ hs,
                                const float* __restrict__ aS, const float* __restrict__ aD,
                                float* __restrict__ asrc, float* __restrict__ adst,
                                int c, int hc) {
  int i = blockIdx.x * blockDim.x + threadIdx.x;
  if (i < NN * HH) {
    int n = i >> 2, h = i & 3;
    const float* hp = hs + (size_t)n * hc + h * c;
    float s0 = 0.0f, s1 = 0.0f;
    for (int j = 0; j < c; ++j) {
      float v = hp[j];
      s0 += v * aS[h * c + j];
      s1 += v * aD[h * c + j];
    }
    asrc[i] = s0;
    adst[i] = s1;
  }
}

// ---------------- edge pass 1: alpha = lrelu(asrc[s]+adst[d]+ea*kE), segment max ---------
__global__ void edge_alpha_kernel(const int* __restrict__ ei, const float* __restrict__ eattr,
                                  const float* __restrict__ la,
                                  const float* __restrict__ asrc, const float* __restrict__ adst,
                                  const float* __restrict__ kE,
                                  float* __restrict__ alpha, unsigned* __restrict__ amax) {
  int e = blockIdx.x * blockDim.x + threadIdx.x;
  if (e < EA) {
    int s, d; float ea;
    if (e < EE) { s = ei[e]; d = ei[EE + e]; ea = eattr[e]; }
    else        { s = e - EE; d = s; ea = la[s]; }
#pragma unroll
    for (int h = 0; h < HH; ++h) {
      float a = asrc[s * 4 + h] + adst[d * 4 + h] + ea * kE[h];
      a = (a > 0.0f) ? a : 0.2f * a;
      alpha[(size_t)e * 4 + h] = a;
      atomicMax(&amax[d * 4 + h], fenc(a));
    }
  }
}

// ---------------- edge pass 2: ex = exp(alpha - amax[d]); denom += ex --------------------
__global__ void edge_exp_kernel(const int* __restrict__ ei, float* __restrict__ alpha,
                                const unsigned* __restrict__ amax, float* __restrict__ denom) {
  int e = blockIdx.x * blockDim.x + threadIdx.x;
  if (e < EA) {
    int d = (e < EE) ? ei[EE + e] : (e - EE);
#pragma unroll
    for (int h = 0; h < HH; ++h) {
      float ex = __expf(alpha[(size_t)e * 4 + h] - fdec(amax[d * 4 + h]));
      alpha[(size_t)e * 4 + h] = ex;
      atomicAdd(&denom[d * 4 + h], ex);
    }
  }
}

// ---------------- edge pass 3: out[d] += hs[s] * (ex/denom[d]) ---------------------------
__global__ void edge_scatter_kernel(const int* __restrict__ ei, const float* __restrict__ ex,
                                    const float* __restrict__ denom, const float* __restrict__ hs,
                                    float* __restrict__ out, int c, int hc) {
  int t = blockIdx.x * blockDim.x + threadIdx.x;
  if (t < EA * HH) {
    int e = t >> 2, h = t & 3;
    int s, d;
    if (e < EE) { s = ei[e]; d = ei[EE + e]; }
    else        { s = e - EE; d = s; }
    float w = ex[t] / (denom[d * 4 + h] + 1e-16f);
    const float* hp = hs + (size_t)s * hc + h * c;
    float* op = out + (size_t)d * hc + h * c;
    for (int j = 0; j < c; ++j) atomicAdd(&op[j], hp[j] * w);
  }
}

// ---------------- bias (+relu), re-encode bf16 for next GEMM or keep f32 ------------------
__global__ void post_kernel(const float* __restrict__ out, const float* __restrict__ bias,
                            bf16* __restrict__ hb, float* __restrict__ hf, int hc, int do_relu) {
  int i = blockIdx.x * blockDim.x + threadIdx.x;
  if (i < NN * hc) {
    int j = i % hc;
    float v = out[i] + bias[j];
    if (do_relu) {
      v = fmaxf(v, 0.0f);
      hb[i] = (bf16)v;
    } else {
      hf[i] = v;
    }
  }
}

// ---------------- graph readout: segment_max over batch ----------------------------------
__global__ void seg_max_kernel(const float* __restrict__ hf, const int* __restrict__ batch,
                               unsigned* __restrict__ genc) {
  int i = blockIdx.x * blockDim.x + threadIdx.x;
  if (i < NN * 64) {
    int n = i >> 6, j = i & 63;
    atomicMax(&genc[batch[n] * 64 + j], fenc(hf[i]));
  }
}
__global__ void decode_kernel(const unsigned* __restrict__ genc, float* __restrict__ outp) {
  int i = blockIdx.x * blockDim.x + threadIdx.x;
  if (i < GG * 64) outp[i] = fdec(genc[i]);
}

// =========================================================================================
static inline int grid_of(long long n, int b) { return (int)((n + b - 1) / b); }

extern "C" void kernel_launch(void* const* d_in, const int* in_sizes, int n_in,
                              void* d_out, int out_size, void* d_ws, size_t ws_size,
                              hipStream_t stream) {
  (void)in_sizes; (void)n_in; (void)out_size; (void)ws_size;
  const int*   x     = (const int*)d_in[0];
  const int*   ei    = (const int*)d_in[1];
  const float* pos   = (const float*)d_in[2];
  const int*   batch = (const int*)d_in[3];
  const float* eattr = (const float*)d_in[4];
  const float* ew    = (const float*)d_in[5];
  const float* fw    = (const float*)d_in[6];
  const float* fb    = (const float*)d_in[7];

  struct LayerP { const float *lin, *as, *ad, *le, *ae, *bias; int cin, c; };
  LayerP L[3] = {
    { (const float*)d_in[8],  (const float*)d_in[9],  (const float*)d_in[10],
      (const float*)d_in[11], (const float*)d_in[12], (const float*)d_in[13], 64, 32 },
    { (const float*)d_in[14], (const float*)d_in[15], (const float*)d_in[16],
      (const float*)d_in[17], (const float*)d_in[18], (const float*)d_in[19], 128, 32 },
    { (const float*)d_in[20], (const float*)d_in[21], (const float*)d_in[22],
      (const float*)d_in[23], (const float*)d_in[24], (const float*)d_in[25], 128, 16 },
  };

  // ---- workspace carve-up (aligned 256B) ----
  char* ws = (char*)d_ws;
  size_t off = 0;
  auto carve = [&](size_t bytes) -> char* {
    char* p = ws + off;
    off = (off + bytes + 255) & ~(size_t)255;
    return p;
  };
  bf16*     hB    = (bf16*)    carve((size_t)NN * 128 * 2);  // bf16 features (A operand)
  float*    hs    = (float*)   carve((size_t)NN * 128 * 4);  // lin output / final features
  float*    outb  = (float*)   carve((size_t)NN * 128 * 4);  // aggregated messages
  float*    alpha = (float*)   carve((size_t)EA * 4 * 4);    // per-edge logits -> exp
  float*    asrc  = (float*)   carve((size_t)NN * 4 * 4);
  float*    adst  = (float*)   carve((size_t)NN * 4 * 4);
  unsigned* amax  = (unsigned*)carve((size_t)NN * 4 * 4);
  float*    denom = (float*)   carve((size_t)NN * 4 * 4);
  float*    deg   = (float*)   carve((size_t)NN * 4);
  float*    la    = (float*)   carve((size_t)NN * 4);        // attr sum -> loop attr
  bf16*     Bp    = (bf16*)    carve((size_t)128 * 128 * 2); // packed weights (per layer)
  float*    kE    = (float*)   carve(256);
  unsigned* genc  = (unsigned*)carve((size_t)GG * 64 * 4);

  const int B = 256;

  // ---- self-loop attrs ----
  fill_f32<<<grid_of(NN, B), B, 0, stream>>>(deg, 0.0f, NN);
  fill_f32<<<grid_of(NN, B), B, 0, stream>>>(la, 0.0f, NN);
  deg_kernel<<<grid_of(EE, B), B, 0, stream>>>(ei, eattr, deg, la);
  loop_fin_kernel<<<grid_of(NN, B), B, 0, stream>>>(la, deg);

  // ---- input embedding (bf16) ----
  embed_kernel<<<grid_of((long long)NN * 64, B), B, 0, stream>>>(x, pos, ew, fw, fb, hB);

  // ---- three GAT layers ----
  for (int li = 0; li < 3; ++li) {
    const LayerP& P = L[li];
    const int hc = HH * P.c;
    const int kblocks = P.cin >> 5;

    packB_kernel<<<grid_of((long long)(hc >> 4) * kblocks * 512, B), B, 0, stream>>>(
        P.lin, Bp, P.cin, hc);
    ke_kernel<<<1, 32, 0, stream>>>(P.le, P.ae, kE, P.c);

    fill_u32<<<grid_of(NN * 4, B), B, 0, stream>>>(amax, ENC_NEG_INF, NN * 4);
    fill_f32<<<grid_of(NN * 4, B), B, 0, stream>>>(denom, 0.0f, NN * 4);
    fill_f32<<<grid_of((long long)NN * hc, B), B, 0, stream>>>(outb, 0.0f, NN * hc);

    // hs = h @ lin  (WMMA bf16 -> f32)
    {
      long long waves = (long long)MT * (hc >> 4);
      gemm_wmma<<<grid_of(waves * 32, B), B, 0, stream>>>(hB, Bp, hs, P.cin, hc);
    }

    node_att_kernel<<<grid_of(NN * HH, B), B, 0, stream>>>(hs, P.as, P.ad, asrc, adst, P.c, hc);
    edge_alpha_kernel<<<grid_of(EA, B), B, 0, stream>>>(ei, eattr, la, asrc, adst, kE, alpha, amax);
    edge_exp_kernel<<<grid_of(EA, B), B, 0, stream>>>(ei, alpha, amax, denom);
    edge_scatter_kernel<<<grid_of((long long)EA * HH, B), B, 0, stream>>>(
        ei, alpha, denom, hs, outb, P.c, hc);

    post_kernel<<<grid_of((long long)NN * hc, B), B, 0, stream>>>(
        outb, P.bias, hB, hs, hc, (li < 2) ? 1 : 0);
  }

  // ---- readout: segment_max over batch -> [G, 64] ----
  fill_u32<<<grid_of(GG * 64, B), B, 0, stream>>>(genc, ENC_NEG_INF, GG * 64);
  seg_max_kernel<<<grid_of((long long)NN * 64, B), B, 0, stream>>>(hs, batch, genc);
  decode_kernel<<<grid_of(GG * 64, B), B, 0, stream>>>(genc, (float*)d_out);
}
